// RelationalReasoningModel_63041529971196
// MI455X (gfx1250) — matrix operand order repaired and checked by reading, not compiled, not run on hardware
//
#include <hip/hip_runtime.h>

// Shapes from the reference
#define BATCH 64
#define NOBJ  49
#define FDIM  512
#define QDIM  128
#define GDIM  256
#define FPDIM 256
#define CDIM  28
#define NROWS (BATCH * NOBJ)   // 3136

typedef __attribute__((ext_vector_type(16))) __bf16 v16bf;
typedef __attribute__((ext_vector_type(8)))  __bf16 v8bf;
typedef __attribute__((ext_vector_type(8)))  float  v8f;

#define XSTRIDE 264   // 256 + 8 bf16 pad -> 4-bank rotation per row, 16B aligned

// ---------------------------------------------------------------- zero S
__global__ __launch_bounds__(256) void rn_zero_kernel(float* __restrict__ S) {
    S[blockIdx.x * 256 + threadIdx.x] = 0.0f;
}

// --------------------------------------------- pack g2_w into WMMA B layout
// Fragment f = ct*8 + ks covers B block W2[k=32ks..32ks+31][n=16ct..16ct+15].
// Lane L<16: N = 16ct+L, elements e=0..15 <-> K = 32ks+e.
// Lane L>=16: N = 16ct+(L-16), elements e <-> K = 32ks+16+e.
__global__ __launch_bounds__(256) void rn_pack_kernel(const float* __restrict__ g2_w,
                                                      unsigned short* __restrict__ pack) {
    int tid  = blockIdx.x * 256 + threadIdx.x;   // 0..4095
    int f    = tid >> 5;                         // 0..127
    int lane = tid & 31;
    int ct   = f >> 3;
    int ks   = f & 7;
    int n     = ct * 16 + (lane & 15);
    int kbase = ks * 32 + ((lane < 16) ? 0 : 16);
    unsigned short* dst = pack + (size_t)(f * 32 + lane) * 16;
    #pragma unroll
    for (int e = 0; e < 16; ++e) {
        __bf16 hv = (__bf16)g2_w[(kbase + e) * GDIM + n];
        dst[e] = __builtin_bit_cast(unsigned short, hv);
    }
}

// ------------------------------------------- qc[b][c] = b1[c] + q[b] @ Wq
__global__ __launch_bounds__(256) void rn_qc_kernel(const float* __restrict__ q,
                                                    const float* __restrict__ g1_w,
                                                    const float* __restrict__ g1_b,
                                                    float* __restrict__ qc) {
    int b = blockIdx.x, c = threadIdx.x;
    __shared__ float sq[QDIM];
    if (c < QDIM) sq[c] = q[b * QDIM + c];
    __syncthreads();
    float acc = g1_b[c];
    #pragma unroll 4
    for (int k = 0; k < QDIM; ++k)
        acc += sq[k] * g1_w[(2 * FDIM + k) * GDIM + c];
    qc[b * GDIM + c] = acc;
}

// --------------------- A'[row] = obj[row]@Wi + qc[b],  B[row] = obj[row]@Wj
// 8 rows per workgroup to amortize g1_w traffic.
__global__ __launch_bounds__(256) void rn_ab_kernel(const float* __restrict__ obj,
                                                    const float* __restrict__ g1_w,
                                                    const float* __restrict__ qc,
                                                    float* __restrict__ Aprime,
                                                    float* __restrict__ Bmat) {
    int r0 = blockIdx.x * 8;
    int t  = threadIdx.x;
    __shared__ float sobj[8][FDIM];   // 16 KB
    for (int idx = t; idx < 8 * FDIM; idx += 256)
        sobj[idx >> 9][idx & (FDIM - 1)] = obj[(size_t)r0 * FDIM + idx];
    __syncthreads();

    int c = t;
    float ai[8], aj[8];
    #pragma unroll
    for (int r = 0; r < 8; ++r) { ai[r] = 0.f; aj[r] = 0.f; }
    for (int k = 0; k < FDIM; ++k) {
        float wi = g1_w[k * GDIM + c];
        float wj = g1_w[(FDIM + k) * GDIM + c];
        #pragma unroll
        for (int r = 0; r < 8; ++r) {
            float o = sobj[r][k];
            ai[r] += o * wi;
            aj[r] += o * wj;
        }
    }
    #pragma unroll
    for (int r = 0; r < 8; ++r) {
        int row = r0 + r;
        int b   = row / NOBJ;
        Aprime[(size_t)row * GDIM + c] = ai[r] + qc[b * GDIM + c];
        Bmat[(size_t)row * GDIM + c]   = aj[r];
    }
}

// ------------------------------------------------ main WMMA pair kernel
// One workgroup per (b,i): X[j] = relu(A'[b,i] + B[b,j]) (64 rows, 49 valid),
// Y = relu(X @ W2 + b2), masked column-sum over j, atomically added into S[b].
__global__ __launch_bounds__(256) void rn_pair_kernel(const float* __restrict__ Aprime,
                                                      const float* __restrict__ Bmat,
                                                      const unsigned short* __restrict__ packW2,
                                                      const float* __restrict__ g2_b,
                                                      float* __restrict__ S) {
    __shared__ __attribute__((aligned(16))) __bf16 sX[64][XSTRIDE];  // ~33 KB
    __shared__ float sA[GDIM];

    int wg = blockIdx.x;            // b*49 + i
    int b  = wg / NOBJ;
    int t  = threadIdx.x;

    sA[t] = Aprime[(size_t)wg * GDIM + t];
    __syncthreads();

    {   // build X tile (rows 49..63 zero)
        int j  = t >> 2;
        int kc = (t & 3) * 64;
        if (j < NOBJ) {
            const float* bm = Bmat + ((size_t)(b * NOBJ + j)) * GDIM + kc;
            #pragma unroll 8
            for (int k = 0; k < 64; ++k) {
                float v = sA[kc + k] + bm[k];
                v = v > 0.f ? v : 0.f;
                sX[j][kc + k] = (__bf16)v;
            }
        } else {
            #pragma unroll 8
            for (int k = 0; k < 64; ++k) sX[j][kc + k] = (__bf16)0.f;
        }
    }
    __syncthreads();

    int wave  = t >> 5;
    int lane  = t & 31;
    int lhalf = (lane < 16) ? 0 : 1;
    int lmod  = lane & 15;

    for (int cti = 0; cti < 2; ++cti) {
        int ct = wave + cti * 8;                       // column tile 0..15
        // keep all 8 K-step B fragments resident in VGPRs
        v16bf bf[8];
        const v16bf* bp = (const v16bf*)packW2 + (size_t)(ct * 8) * 32 + lane;
        #pragma unroll
        for (int ks = 0; ks < 8; ++ks) bf[ks] = bp[(size_t)ks * 32];

        float bias   = g2_b[ct * 16 + lmod];
        float colsum = 0.f;

        #pragma unroll
        for (int rt = 0; rt < 4; ++rt) {
            v8f acc = {};
            #pragma unroll
            for (int ks = 0; ks < 8; ++ks) {
                int m  = rt * 16 + lmod;
                int kb = ks * 32 + lhalf * 16;
                v8bf lo = *(const v8bf*)&sX[m][kb];
                v8bf hi = *(const v8bf*)&sX[m][kb + 8];
                v16bf a;
                #pragma unroll
                for (int e = 0; e < 8; ++e) { a[e] = lo[e]; a[e + 8] = hi[e]; }
                acc = __builtin_amdgcn_wmma_f32_16x16x32_bf16(
                          false, a, false, bf[ks], (short)0, acc, false, false);
            }
            // bias + relu + masked column accumulate (C layout: M = r + 8*lhalf)
            #pragma unroll
            for (int r = 0; r < 8; ++r) {
                int m   = rt * 16 + r + lhalf * 8;
                float y = acc[r] + bias;
                y = y > 0.f ? y : 0.f;
                if (m < NOBJ) colsum += y;
            }
        }
        __hip_atomic_fetch_add(&S[b * GDIM + ct * 16 + lmod], colsum,
                               __ATOMIC_RELAXED, __HIP_MEMORY_SCOPE_AGENT);
    }
}

// ------------------------------------- subtract the 49 diagonal (i==i) pairs
__global__ __launch_bounds__(256) void rn_diag_kernel(const float* __restrict__ Aprime,
                                                      const float* __restrict__ Bmat,
                                                      const float* __restrict__ g2_w,
                                                      const float* __restrict__ g2_b,
                                                      float* __restrict__ S) {
    int b = blockIdx.x, n = threadIdx.x;
    __shared__ float sx[GDIM];
    float sub  = 0.f;
    float bias = g2_b[n];
    for (int i = 0; i < NOBJ; ++i) {
        size_t row = (size_t)(b * NOBJ + i);
        float x = Aprime[row * GDIM + n] + Bmat[row * GDIM + n];
        x = x > 0.f ? x : 0.f;
        sx[n] = (float)(__bf16)x;          // match main path's bf16 rounding
        __syncthreads();
        float acc = 0.f;
        #pragma unroll 4
        for (int k = 0; k < GDIM; ++k)
            acc += sx[k] * (float)(__bf16)g2_w[k * GDIM + n];
        float y = acc + bias;
        sub += (y > 0.f ? y : 0.f);
        __syncthreads();
    }
    S[b * GDIM + n] -= sub;
}

// --------------------------------------------------- head: S -> h2 -> out
__global__ __launch_bounds__(256) void rn_head_kernel(const float* __restrict__ S,
                                                      const float* __restrict__ f1_w,
                                                      const float* __restrict__ f1_b,
                                                      const float* __restrict__ f2_w,
                                                      const float* __restrict__ f2_b,
                                                      float* __restrict__ out) {
    int b = blockIdx.x, n = threadIdx.x;
    __shared__ float sS[GDIM];
    __shared__ float sH[FPDIM];
    sS[n] = S[b * GDIM + n];
    __syncthreads();
    float acc = f1_b[n];
    #pragma unroll 4
    for (int k = 0; k < GDIM; ++k) acc += sS[k] * f1_w[k * FPDIM + n];
    sH[n] = acc > 0.f ? acc : 0.f;
    __syncthreads();
    if (n < CDIM) {
        float o = f2_b[n];
        #pragma unroll 4
        for (int k = 0; k < FPDIM; ++k) o += sH[k] * f2_w[k * CDIM + n];
        out[b * CDIM + n] = o;
    }
}

extern "C" void kernel_launch(void* const* d_in, const int* in_sizes, int n_in,
                              void* d_out, int out_size, void* d_ws, size_t ws_size,
                              hipStream_t stream) {
    const float* obj  = (const float*)d_in[0];
    const float* q    = (const float*)d_in[1];
    const float* g1_w = (const float*)d_in[2];
    const float* g1_b = (const float*)d_in[3];
    const float* g2_w = (const float*)d_in[4];
    const float* g2_b = (const float*)d_in[5];
    const float* f1_w = (const float*)d_in[6];
    const float* f1_b = (const float*)d_in[7];
    const float* f2_w = (const float*)d_in[8];
    const float* f2_b = (const float*)d_in[9];
    float* out = (float*)d_out;

    // workspace layout (~6.7 MB)
    float* Aprime = (float*)d_ws;                 // 3136*256
    float* Bmat   = Aprime + (size_t)NROWS * GDIM; // 3136*256
    float* S      = Bmat   + (size_t)NROWS * GDIM; // 64*256
    float* qc     = S      + (size_t)BATCH * GDIM; // 64*256
    unsigned short* pack = (unsigned short*)(qc + (size_t)BATCH * GDIM); // 65536 bf16

    rn_zero_kernel<<<BATCH, 256, 0, stream>>>(S);
    rn_pack_kernel<<<16, 256, 0, stream>>>(g2_w, pack);
    rn_qc_kernel<<<BATCH, 256, 0, stream>>>(q, g1_w, g1_b, qc);
    rn_ab_kernel<<<NROWS / 8, 256, 0, stream>>>(obj, g1_w, qc, Aprime, Bmat);
    rn_pair_kernel<<<NROWS, 256, 0, stream>>>(Aprime, Bmat, pack, g2_b, S);
    rn_diag_kernel<<<BATCH, 256, 0, stream>>>(Aprime, Bmat, g2_w, g2_b, S);
    rn_head_kernel<<<BATCH, 256, 0, stream>>>(S, f1_w, f1_b, f2_w, f2_b, out);
}
